// _LSTMBlock_1872605741706
// MI455X (gfx1250) — matrix operand order repaired and checked by reading, not compiled, not run on hardware
//
#include <hip/hip_runtime.h>
#include <hip/hip_bf16.h>
#include <stdint.h>

#define B_ROWS 8192
#define DIM_H  1024
#define KTOT   2048   // D + H concatenated along K

typedef __attribute__((ext_vector_type(16))) __bf16 v16bf;
typedef __attribute__((ext_vector_type(4)))  __bf16 v4bf;
typedef __attribute__((ext_vector_type(8)))  float  v8f;
typedef __attribute__((ext_vector_type(4)))  float  v4f;
typedef __attribute__((ext_vector_type(4)))  unsigned int v4u;

union FragU { v4u q[2]; v16bf v; };

// A fragment (16x32 bf16): lane L<16 -> row M=L, elems[0..7]=K+0..7, elems[8..15]=K+16..23
//                          lane L>=16 -> row M=L-16, elems[0..7]=K+8..15, elems[8..15]=K+24..31
__device__ __forceinline__ v16bf load_a_frag(const __bf16* p) {
  FragU f;
  f.q[0] = *(const v4u*)(p);        // 16B: K+0..7 (or +8..15)
  f.q[1] = *(const v4u*)(p + 16);   // 16B: K+16..23 (or +24..31)
  return f.v;
}
// B fragment (32x16 bf16): lane L<16 -> col N=L, elems[0..7]=K+0..7, elems[8..15]=K+16..23
//                          lane L>=16 -> col N=L-16, elems[0..7]=K+8..15, elems[8..15]=K+24..31
__device__ __forceinline__ v16bf load_b_frag(const __bf16* p) {
  FragU f;
  f.q[0] = *(const v4u*)(p);
  f.q[1] = *(const v4u*)(p + 16);
  return f.v;
}

__device__ __forceinline__ float fsig(float x)  { return 1.0f / (1.0f + __expf(-x)); }
__device__ __forceinline__ float ftanh(float x) {
  float e = __expf(-2.0f * x);
  return (1.0f - e) / (1.0f + e);
}

// ---------------- Pack kernels (f32 -> bf16, concat layouts) ----------------

// xh[row][k], k<1024 from x, k>=1024 from hidden. Vectorized by 4.
__global__ void pack_xh_kernel(const float* __restrict__ x,
                               const float* __restrict__ hidden,
                               __bf16* __restrict__ xh) {
  size_t idx = (size_t)blockIdx.x * blockDim.x + threadIdx.x;
  size_t e = idx * 4;
  if (e >= (size_t)B_ROWS * KTOT) return;
  int row = (int)(e >> 11);
  int k   = (int)(e & (KTOT - 1));
  const float* src = (k < 1024) ? (x + (size_t)row * 1024 + k)
                                : (hidden + (size_t)row * 1024 + (k - 1024));
  v4f vf = *(const v4f*)src;
  v4bf o = __builtin_convertvector(vf, v4bf);
  *(v4bf*)(xh + e) = o;
}

// Wp[g][n][k]: g in {f,i,c,o}; k<1024 from Wg_x[n][k], else Wg_h[n][k-1024]
__global__ void pack_w_kernel(const float* __restrict__ Wfx, const float* __restrict__ Wfh,
                              const float* __restrict__ Wix, const float* __restrict__ Wih,
                              const float* __restrict__ Wcx, const float* __restrict__ Wch,
                              const float* __restrict__ Wox, const float* __restrict__ Woh,
                              __bf16* __restrict__ Wp) {
  size_t idx = (size_t)blockIdx.x * blockDim.x + threadIdx.x;
  size_t e = idx * 4;
  if (e >= (size_t)4 * DIM_H * KTOT) return;
  int g = (int)(e >> 21);                 // 1024*2048 = 2^21 per gate
  size_t r = e & ((1u << 21) - 1);
  int n = (int)(r >> 11);
  int k = (int)(r & (KTOT - 1));
  const float* Wx = (g == 0) ? Wfx : (g == 1) ? Wix : (g == 2) ? Wcx : Wox;
  const float* Wh = (g == 0) ? Wfh : (g == 1) ? Wih : (g == 2) ? Wch : Woh;
  const float* src = (k < 1024) ? (Wx + (size_t)n * 1024 + k)
                                : (Wh + (size_t)n * 1024 + (k - 1024));
  v4f vf = *(const v4f*)src;
  v4bf o = __builtin_convertvector(vf, v4bf);
  *(v4bf*)(Wp + e) = o;
}

// bcomb[g][n] = b{g}_x[n] + b{g}_h[n]
__global__ void pack_bias_kernel(const float* __restrict__ bfx, const float* __restrict__ bfh,
                                 const float* __restrict__ bix, const float* __restrict__ bih,
                                 const float* __restrict__ bcx, const float* __restrict__ bch,
                                 const float* __restrict__ box, const float* __restrict__ boh,
                                 float* __restrict__ bc) {
  int i = blockIdx.x * blockDim.x + threadIdx.x;
  if (i >= 4 * DIM_H) return;
  int g = i >> 10, n = i & (DIM_H - 1);
  const float* bx = (g == 0) ? bfx : (g == 1) ? bix : (g == 2) ? bcx : box;
  const float* bh = (g == 0) ? bfh : (g == 1) ? bih : (g == 2) ? bch : boh;
  bc[i] = bx[n] + bh[n];
}

// ---------------- Fused 4-gate WMMA GEMM + LSTM elementwise ----------------
// Grid: (B/64, H/128). Block: 256 threads = 8 waves (2x4). Wave tile: 32x32 per gate.
__global__ __launch_bounds__(256, 1) void lstm_wmma_kernel(
    const __bf16* __restrict__ xh,     // [B][2048]
    const __bf16* __restrict__ Wp,     // [4][1024][2048]
    const float*  __restrict__ bcomb,  // [4][1024]
    const float*  __restrict__ cell,   // [B][1024]
    float* __restrict__ c_out,         // [B][1024]
    float* __restrict__ h_out)         // [B][1024]
{
  const int lane = threadIdx.x & 31;
  const int wave = threadIdx.x >> 5;
  const int m0 = blockIdx.x * 64  + (wave >> 2) * 32;
  const int n0 = blockIdx.y * 128 + (wave & 3) * 32;

  const int fr = lane & 15;   // row (A) / col (B) within fragment
  const int hi = lane >> 4;   // K-half selector

  v8f acc[4][2][2];
#pragma unroll
  for (int g = 0; g < 4; ++g)
#pragma unroll
    for (int mi = 0; mi < 2; ++mi)
#pragma unroll
      for (int ni = 0; ni < 2; ++ni)
        acc[g][mi][ni] = {};

  const __bf16* aRow = xh + (size_t)(m0 + fr) * KTOT + hi * 8;
  const __bf16* bRow = Wp + (size_t)(n0 + fr) * KTOT + hi * 8;

  for (int k0 = 0; k0 < KTOT; k0 += 32) {
    if (k0 + 32 < KTOT) {
      __builtin_prefetch(aRow + k0 + 32, 0, 3);                     // global_prefetch_b8
      __builtin_prefetch(aRow + (size_t)16 * KTOT + k0 + 32, 0, 3);
    }
    v16bf a0 = load_a_frag(aRow + k0);
    v16bf a1 = load_a_frag(aRow + (size_t)16 * KTOT + k0);
#pragma unroll
    for (int g = 0; g < 4; ++g) {
      const __bf16* bg = bRow + (size_t)g * ((size_t)DIM_H * KTOT) + k0;
      v16bf b0 = load_b_frag(bg);
      v16bf b1 = load_b_frag(bg + (size_t)16 * KTOT);
      acc[g][0][0] = __builtin_amdgcn_wmma_f32_16x16x32_bf16(false, a0, false, b0, (short)0, acc[g][0][0], false, false);
      acc[g][0][1] = __builtin_amdgcn_wmma_f32_16x16x32_bf16(false, a0, false, b1, (short)0, acc[g][0][1], false, false);
      acc[g][1][0] = __builtin_amdgcn_wmma_f32_16x16x32_bf16(false, a1, false, b0, (short)0, acc[g][1][0], false, false);
      acc[g][1][1] = __builtin_amdgcn_wmma_f32_16x16x32_bf16(false, a1, false, b1, (short)0, acc[g][1][1], false, false);
    }
  }

  // Epilogue: C/D layout -> lane<16: N=lane, M=v; lane>=16: N=lane-16, M=v+8
  const int rowOff = hi * 8;
#pragma unroll
  for (int mi = 0; mi < 2; ++mi) {
#pragma unroll
    for (int ni = 0; ni < 2; ++ni) {
      const int col = n0 + ni * 16 + fr;
      const float bF = bcomb[0 * DIM_H + col];
      const float bI = bcomb[1 * DIM_H + col];
      const float bC = bcomb[2 * DIM_H + col];
      const float bO = bcomb[3 * DIM_H + col];
#pragma unroll
      for (int v = 0; v < 8; ++v) {
        const int row = m0 + mi * 16 + rowOff + v;
        const size_t off = (size_t)row * DIM_H + col;
        const float f  = fsig (acc[0][mi][ni][v] + bF);
        const float ig = fsig (acc[1][mi][ni][v] + bI);
        const float cc = ftanh(acc[2][mi][ni][v] + bC);
        const float o  = fsig (acc[3][mi][ni][v] + bO);
        const float cn = cell[off] * f + ig * cc;
        c_out[off] = cn;
        h_out[off] = o * ftanh(cn);
      }
    }
  }
}

// ---------------- Launcher ----------------
extern "C" void kernel_launch(void* const* d_in, const int* in_sizes, int n_in,
                              void* d_out, int out_size, void* d_ws, size_t ws_size,
                              hipStream_t stream) {
  // d_in order: 0 cell, 1 hidden, 2 x, 3 Wf_x, 4 bf_x, 5 Wf_h, 6 bf_h,
  //             7 Wi_x, 8 bi_x, 9 Wi_h, 10 bi_h, 11 Wc_x, 12 bc_x, 13 Wc_h, 14 bc_h,
  //             15 Wo_x, 16 bo_x, 17 Wo_h, 18 bo_h
  const float* cell   = (const float*)d_in[0];
  const float* hidden = (const float*)d_in[1];
  const float* x      = (const float*)d_in[2];

  // Workspace layout
  const size_t xh_bytes = (size_t)B_ROWS * KTOT * sizeof(__bf16);        // 32 MB
  const size_t wp_bytes = (size_t)4 * DIM_H * KTOT * sizeof(__bf16);     // 16 MB
  __bf16* xh = (__bf16*)d_ws;
  __bf16* Wp = (__bf16*)((char*)d_ws + xh_bytes);
  float*  bc = (float*)((char*)d_ws + xh_bytes + wp_bytes);

  float* c_out = (float*)d_out;
  float* h_out = c_out + (size_t)B_ROWS * DIM_H;

  {
    size_t nvec = (size_t)B_ROWS * KTOT / 4;
    pack_xh_kernel<<<(unsigned)((nvec + 255) / 256), 256, 0, stream>>>(x, hidden, xh);
  }
  {
    size_t nvec = (size_t)4 * DIM_H * KTOT / 4;
    pack_w_kernel<<<(unsigned)((nvec + 255) / 256), 256, 0, stream>>>(
        (const float*)d_in[3],  (const float*)d_in[5],   // Wf_x, Wf_h
        (const float*)d_in[7],  (const float*)d_in[9],   // Wi_x, Wi_h
        (const float*)d_in[11], (const float*)d_in[13],  // Wc_x, Wc_h
        (const float*)d_in[15], (const float*)d_in[17],  // Wo_x, Wo_h
        Wp);
  }
  pack_bias_kernel<<<(4 * DIM_H + 255) / 256, 256, 0, stream>>>(
      (const float*)d_in[4],  (const float*)d_in[6],
      (const float*)d_in[8],  (const float*)d_in[10],
      (const float*)d_in[12], (const float*)d_in[14],
      (const float*)d_in[16], (const float*)d_in[18],
      bc);

  dim3 grid(B_ROWS / 64, DIM_H / 128);
  lstm_wmma_kernel<<<grid, 256, 0, stream>>>(xh, Wp, bc, cell, c_out, h_out);
}
